// PointNet2Encoder_62414464746069
// MI455X (gfx1250) — matrix operand order repaired and checked by reading, not compiled, not run on hardware
//
#include <hip/hip_runtime.h>
#include <hip/hip_bf16.h>
#include <utility>

// ============================================================================
// PointNet++ MSG encoder for MI455X (gfx1250, wave32, WMMA).
// All MLP layers run as v_wmma_f32_16x16x32_f16 GEMMs with BN+ReLU fused in
// the epilogue. Each wave computes a 16x32 output tile (2 accumulators
// sharing one A fragment), with software-pipelined fragment loads so the
// next K-step's VMEM is in flight while the current WMMAs execute.
// Activations f16, accumulation f32. FPS / ball-query are scan kernels
// (LDS-resident distances, wave32 ballot compaction).
// Workspace requirement: ~280 MB (two 134MB ping-pong activation arenas for
// the largest stage, SA1 scale2: 524288 rows x 128 ch f16).
// ============================================================================

typedef __attribute__((ext_vector_type(16))) _Float16 v16h;
typedef __attribute__((ext_vector_type(8)))  float    v8f;

#define EPS_BN 1e-5f

static __device__ __forceinline__ float f16tof(_Float16 h) { return (float)h; }

// ---------------------------------------------------------------------------
// Transpose input (B,3,N) -> (B,N,3)
// ---------------------------------------------------------------------------
__global__ void transpose_xyz(const float* __restrict__ in, float* __restrict__ out,
                              int Bn, int N) {
    int t = blockIdx.x * blockDim.x + threadIdx.x;
    if (t >= Bn * N) return;
    int b = t / N, n = t % N;
    out[(size_t)t * 3 + 0] = in[(size_t)b * 3 * N + 0 * N + n];
    out[(size_t)t * 3 + 1] = in[(size_t)b * 3 * N + 1 * N + n];
    out[(size_t)t * 3 + 2] = in[(size_t)b * 3 * N + 2 * N + n];
}

// ---------------------------------------------------------------------------
// Farthest point sampling. One workgroup (1024 threads) per batch element.
// Distance array lives in LDS (max N=8192 -> 32KB) + 8KB reduction scratch.
// Matches jax.lax.scan semantics: emit `far` first, then update.
// ---------------------------------------------------------------------------
__global__ __launch_bounds__(1024) void fps_kernel(const float* __restrict__ xyz,
                                                   int N, int np, int* __restrict__ out) {
    __shared__ float sd[8192];
    __shared__ float rv[1024];
    __shared__ int   ri[1024];
    int b = blockIdx.x, tid = threadIdx.x, bs = blockDim.x;
    const float* X = xyz + (size_t)b * N * 3;
    for (int j = tid; j < N; j += bs) sd[j] = 1e10f;
    __syncthreads();
    int far = 0;
    for (int it = 0; it < np; ++it) {
        if (tid == 0) out[b * np + it] = far;
        float cx = X[far * 3 + 0], cy = X[far * 3 + 1], cz = X[far * 3 + 2];
        float bv = -1.0f; int bi = 0x7fffffff;
        for (int j = tid; j < N; j += bs) {
            float dx = X[j * 3 + 0] - cx;
            float dy = X[j * 3 + 1] - cy;
            float dz = X[j * 3 + 2] - cz;
            float d  = dx * dx + dy * dy + dz * dz;
            float nd = fminf(sd[j], d);
            sd[j] = nd;
            if (nd > bv || (nd == bv && j < bi)) { bv = nd; bi = j; }
        }
        rv[tid] = bv; ri[tid] = bi;
        __syncthreads();
        for (int s = bs >> 1; s > 0; s >>= 1) {
            if (tid < s) {
                if (rv[tid + s] > rv[tid] ||
                    (rv[tid + s] == rv[tid] && ri[tid + s] < ri[tid])) {
                    rv[tid] = rv[tid + s]; ri[tid] = ri[tid + s];
                }
            }
            __syncthreads();
        }
        far = ri[0];
        __syncthreads();
    }
}

// Gather sampled centers: nxyz[b,s,:] = xyz[b, fidx[b,s], :]
__global__ void gather_centers(const float* __restrict__ xyz, const int* __restrict__ fidx,
                               float* __restrict__ nxyz, int N, int S, int total) {
    int t = blockIdx.x * blockDim.x + threadIdx.x;
    if (t >= total) return;
    int b = t / S;
    int j = fidx[t];
    const float* p = xyz + ((size_t)b * N + j) * 3;
    nxyz[(size_t)t * 3 + 0] = p[0];
    nxyz[(size_t)t * 3 + 1] = p[1];
    nxyz[(size_t)t * 3 + 2] = p[2];
}

// ---------------------------------------------------------------------------
// Ball query: one wave per center, ordered wave32 ballot compaction.
// Selects the first K indices (ascending) with sq-dist <= r^2, fills the
// remainder with the first selected index (reference semantics).
// ---------------------------------------------------------------------------
__global__ void query_ball_kernel(const float* __restrict__ xyz, const float* __restrict__ nxyz,
                                  int N, int S, float r2, int K, int* __restrict__ idx,
                                  int ncent) {
    int wid  = (blockIdx.x * blockDim.x + threadIdx.x) >> 5;
    int lane = threadIdx.x & 31;
    if (wid >= ncent) return;
    int b = wid / S;
    const float* X = xyz + (size_t)b * N * 3;
    float cx = nxyz[(size_t)wid * 3 + 0];
    float cy = nxyz[(size_t)wid * 3 + 1];
    float cz = nxyz[(size_t)wid * 3 + 2];
    int* out = idx + (size_t)wid * K;
    int cnt = 0, firstj = 0;
    for (int base = 0; base < N && cnt < K; base += 32) {
        int j = base + lane;
        bool inr = false;
        if (j < N) {
            float dx = X[j * 3 + 0] - cx;
            float dy = X[j * 3 + 1] - cy;
            float dz = X[j * 3 + 2] - cz;
            inr = (dx * dx + dy * dy + dz * dz) <= r2;
        }
        unsigned mask = (unsigned)__ballot(inr);
        if (mask) {
            if (cnt == 0) firstj = base + (__ffs(mask) - 1);
            int pos = cnt + __popc(mask & (lane ? ((1u << lane) - 1u) : 0u));
            if (inr && pos < K) out[pos] = j;
            cnt += __popc(mask);
        }
    }
    for (int t = cnt + lane; t < K; t += 32) out[t] = firstj;
}

// ---------------------------------------------------------------------------
// Grouping gathers -> f16 GEMM A-matrices (row-major [rows, Kp], zero-padded)
// ---------------------------------------------------------------------------
// SA1: channels = relative xyz only (3 -> Kp=32). One thread per row.
__global__ void gather_group_xyz(const float* __restrict__ xyz, const float* __restrict__ nxyz,
                                 const int* __restrict__ idx, _Float16* __restrict__ G,
                                 int N, int S, int K, int rows) {
    int row = blockIdx.x * blockDim.x + threadIdx.x;
    if (row >= rows) return;
    int c = row / K;           // center = b*S + s
    int b = c / S;
    int j = idx[row];
    const float* p = xyz + ((size_t)b * N + j) * 3;
    const float* q = nxyz + (size_t)c * 3;
    _Float16 vals[32];
#pragma unroll
    for (int i = 0; i < 32; ++i) vals[i] = (_Float16)0.0f;
    vals[0] = (_Float16)(p[0] - q[0]);
    vals[1] = (_Float16)(p[1] - q[1]);
    vals[2] = (_Float16)(p[2] - q[2]);
    uint4* dst = (uint4*)(G + (size_t)row * 32);
    const uint4* src = (const uint4*)vals;
    dst[0] = src[0]; dst[1] = src[1]; dst[2] = src[2]; dst[3] = src[3];
}

// SA2: channels = [Cfeat features, rel xyz, zero-pad] (ref concat order).
// One wave per row.
__global__ void gather_group_feat(const float* __restrict__ xyz, const float* __restrict__ nxyz,
                                  const _Float16* __restrict__ feat, const int* __restrict__ idx,
                                  _Float16* __restrict__ G, int Nsrc, int S, int K,
                                  int Cfeat, int Kp, int rows) {
    int wid  = (blockIdx.x * blockDim.x + threadIdx.x) >> 5;
    int lane = threadIdx.x & 31;
    if (wid >= rows) return;
    int cI = wid / K;
    int b  = cI / S;
    int j  = idx[wid];
    const _Float16* f = feat + ((size_t)b * Nsrc + j) * Cfeat;
    _Float16* g = G + (size_t)wid * Kp;
    for (int c = lane; c < Cfeat; c += 32) g[c] = f[c];
    const float* p = xyz + ((size_t)b * Nsrc + j) * 3;
    const float* q = nxyz + (size_t)cI * 3;
    for (int t = Cfeat + lane; t < Kp; t += 32) {
        int r = t - Cfeat;
        float v = 0.0f;
        if (r == 0) v = p[0] - q[0];
        else if (r == 1) v = p[1] - q[1];
        else if (r == 2) v = p[2] - q[2];
        g[t] = (_Float16)v;
    }
}

// SA3 global stage: channels = [xyz(3), feat(Cfeat), pad]. One wave per row.
__global__ void gather_sa3(const float* __restrict__ l2xyz, const _Float16* __restrict__ l2feat,
                           _Float16* __restrict__ G, int rows, int Cfeat, int Kp) {
    int wid  = (blockIdx.x * blockDim.x + threadIdx.x) >> 5;
    int lane = threadIdx.x & 31;
    if (wid >= rows) return;
    _Float16* g = G + (size_t)wid * Kp;
    for (int c = lane; c < Kp; c += 32) {
        float v;
        if (c < 3)              v = l2xyz[(size_t)wid * 3 + c];
        else if (c < 3 + Cfeat) v = f16tof(l2feat[(size_t)wid * Cfeat + (c - 3)]);
        else                    v = 0.0f;
        g[c] = (_Float16)v;
    }
}

// ---------------------------------------------------------------------------
// Weight conversion: fp32 W[O,C] -> f16 Wh[O,Kp] row-major, zero K padding.
// ---------------------------------------------------------------------------
__global__ void conv_weight(const float* __restrict__ W, _Float16* __restrict__ Wh,
                            int O, int C, int Kp) {
    int t = blockIdx.x * blockDim.x + threadIdx.x;
    if (t >= O * Kp) return;
    int o = t / Kp, k = t % Kp;
    Wh[t] = (_Float16)(k < C ? W[o * C + k] : 0.0f);
}

// ---------------------------------------------------------------------------
// WMMA GEMM + fused BN + ReLU.
//   Out[m,n] = relu( gamma[n]*inv_sqrt(1+eps) * (A@Wh^T + bias[n]) + beta[n] )
// A: [M,Kp] f16 row-major. Wh: [O,Kp] f16 row-major (so B = Wh^T is [Kp,O]).
// Block = 256 threads = 8 waves; each wave owns a 16x32 output tile (two
// v_wmma accumulators sharing one A fragment). Fragment loads are software
// pipelined (next K-step issued before current WMMAs, last step peeled) so
// s_wait_loadcnt overlaps VMEM with the matrix pipe. Fragment layouts per
// CDNA5 ISA §7.12.2.
// ---------------------------------------------------------------------------
union AFrag { v16h v; unsigned u[8]; };

static __device__ __forceinline__ void load_frags(
        const _Float16* arow, const _Float16* b0row, const _Float16* b1row,
        int k0, int half, AFrag& a, AFrag& b0, AFrag& b1) {
#pragma unroll
    for (int v = 0; v < 8; ++v) {
        // A 16x32 f16 fragment: VGPR v holds K pair at
        //   ((v>>2)<<4) + (half<<3) + ((v&3)<<1)          (ISA §7.12.2)
        int ka = k0 + ((v >> 2) << 4) + (half << 3) + ((v & 3) << 1);
        a.u[v] = *(const unsigned*)(arow + ka);
        // B 32x16 f16 fragment: column = lane&15, VGPR v holds K pair at
        //   ((v>>2)<<4) + (half<<3) + ((v&3)<<1)
        int kb = k0 + ((v >> 2) << 4) + (half << 3) + ((v & 3) << 1);
        b0.u[v] = *(const unsigned*)(b0row + kb);
        b1.u[v] = *(const unsigned*)(b1row + kb);
    }
}

__global__ __launch_bounds__(256) void gemm_bn_relu(
        const _Float16* __restrict__ A, const _Float16* __restrict__ Wh,
        const float* __restrict__ bias, const float* __restrict__ gamma,
        const float* __restrict__ beta, _Float16* __restrict__ Out,
        int M, int Kp, int O) {
    int w    = threadIdx.x >> 5;
    int lane = threadIdx.x & 31;
    int m0   = blockIdx.x * 128 + w * 16;
    int n0   = blockIdx.y * 32;
    if (m0 >= M) return;                       // wave-uniform: EXEC stays all-1s
    int half = lane >> 4;
    int mrow  = m0 + (lane & 15);
    int ncol0 = n0 + (lane & 15);
    int ncol1 = ncol0 + 16;
    const _Float16* arow  = A  + (size_t)mrow  * Kp;
    const _Float16* b0row = Wh + (size_t)ncol0 * Kp;
    const _Float16* b1row = Wh + (size_t)ncol1 * Kp;

    v8f acc0 = {}, acc1 = {};
    AFrag a, b0, b1, an, b0n, b1n;
    load_frags(arow, b0row, b1row, 0, half, a, b0, b1);
    int k0 = 0;
    for (; k0 + 32 < Kp; k0 += 32) {
        // Speculative WGP-scope prefetch two K-steps ahead of the A stream;
        // past-the-end prefetches are silently dropped (ISA §10.5).
        __builtin_prefetch(arow + k0 + 64, 0, 3);
        load_frags(arow, b0row, b1row, k0 + 32, half, an, b0n, b1n);
        acc0 = __builtin_amdgcn_wmma_f32_16x16x32_f16(
                   false, a.v, false, b0.v, (short)0, acc0, false, false);
        acc1 = __builtin_amdgcn_wmma_f32_16x16x32_f16(
                   false, a.v, false, b1.v, (short)0, acc1, false, false);
        a = an; b0 = b0n; b1 = b1n;
    }
    acc0 = __builtin_amdgcn_wmma_f32_16x16x32_f16(
               false, a.v, false, b0.v, (short)0, acc0, false, false);
    acc1 = __builtin_amdgcn_wmma_f32_16x16x32_f16(
               false, a.v, false, b1.v, (short)0, acc1, false, false);

    const float invs = rsqrtf(1.0f + EPS_BN);
    float al0 = gamma[ncol0] * invs, bb0 = bias[ncol0], bt0 = beta[ncol0];
    float al1 = gamma[ncol1] * invs, bb1 = bias[ncol1], bt1 = beta[ncol1];
#pragma unroll
    for (int r = 0; r < 8; ++r) {
        int m = m0 + half * 8 + r;             // C layout: VGPR r -> M = half*8+r
        float x0 = al0 * (acc0[r] + bb0) + bt0;
        float x1 = al1 * (acc1[r] + bb1) + bt1;
        x0 = x0 > 0.0f ? x0 : 0.0f;
        x1 = x1 > 0.0f ? x1 : 0.0f;
        Out[(size_t)m * O + ncol0] = (_Float16)x0;
        Out[(size_t)m * O + ncol1] = (_Float16)x1;
    }
}

// ---------------------------------------------------------------------------
// Max-pool over the K (neighbor) axis; writes into feature buffer at coff.
// ---------------------------------------------------------------------------
__global__ void pool_max_k(const _Float16* __restrict__ X, _Float16* __restrict__ F,
                           int K, int O, int Ctot, int coff, int ncent) {
    int t = blockIdx.x * blockDim.x + threadIdx.x;
    if (t >= ncent * O) return;
    int c = t / O, o = t % O;
    const _Float16* x = X + (size_t)c * K * O + o;
    float m = -3.4e38f;
    for (int kk = 0; kk < K; ++kk) {
        float v = f16tof(x[(size_t)kk * O]);
        m = v > m ? v : m;
    }
    F[(size_t)c * Ctot + coff + o] = (_Float16)m;
}

__global__ void zero_f16(_Float16* __restrict__ p, int n) {
    int t = blockIdx.x * blockDim.x + threadIdx.x;
    if (t < n) p[t] = (_Float16)0.0f;
}

__global__ void copy_out_f32(const _Float16* __restrict__ src, float* __restrict__ dst, int n) {
    int t = blockIdx.x * blockDim.x + threadIdx.x;
    if (t < n) dst[t] = f16tof(src[t]);
}

// ===========================================================================
// Host orchestration
// ===========================================================================
namespace {
struct Layer { const float *W, *b, *g, *be; int O, C; };
static inline int div_up(int a, int b) { return (a + b - 1) / b; }
static inline int pad32(int c) { return (c + 31) & ~31; }
}

extern "C" void kernel_launch(void* const* d_in, const int* in_sizes, int n_in,
                              void* d_out, int out_size, void* d_ws, size_t ws_size,
                              hipStream_t stream) {
    (void)in_sizes; (void)n_in; (void)out_size; (void)ws_size;
    const int Bn = 8, N = 8192;
    const int S1 = 512, S2 = 128;

    // ---- parse inputs (setup_inputs dict order) ----
    const float* input = (const float*)d_in[0];
    int pi = 1;
    auto nextLayer = [&](int O, int C) {
        Layer L;
        L.W  = (const float*)d_in[pi++];
        L.b  = (const float*)d_in[pi++];
        L.g  = (const float*)d_in[pi++];
        L.be = (const float*)d_in[pi++];
        L.O = O; L.C = C;
        return L;
    };
    const int w1[3][3] = {{32,32,64},{64,64,128},{64,96,128}};
    const int w2[3][3] = {{64,64,128},{128,128,256},{128,128,256}};
    const int k1[3] = {16,32,128};       const float r1[3] = {0.1f,0.2f,0.4f};
    const int k2[3] = {32,64,128};       const float r2v[3]= {0.2f,0.4f,0.8f};
    const int C1 = 320, C2 = 640;        // concat widths after SA1 / SA2

    Layer sa1[3][3], sa2[3][3], sa3[3], head[2];
    for (int s = 0; s < 3; ++s) { int c = 3;      for (int l = 0; l < 3; ++l) { sa1[s][l] = nextLayer(w1[s][l], c); c = w1[s][l]; } }
    for (int s = 0; s < 3; ++s) { int c = C1 + 3; for (int l = 0; l < 3; ++l) { sa2[s][l] = nextLayer(w2[s][l], c); c = w2[s][l]; } }
    { int c = C2 + 3; const int w3[3] = {256,512,1024};
      for (int l = 0; l < 3; ++l) { sa3[l] = nextLayer(w3[l], c); c = w3[l]; } }
    head[0] = nextLayer(256, 1024);
    head[1] = nextLayer(128, 256);

    // ---- workspace carve-up (bump allocator, 256B aligned) ----
    char* wsp = (char*)d_ws; size_t off = 0;
    auto alloc = [&](size_t bytes) -> void* {
        void* p = wsp + off;
        off = (off + bytes + 255) & ~(size_t)255;
        return p;
    };
    float*     xyz_t   = (float*)    alloc((size_t)Bn * N * 3 * 4);
    int*       fpsIdx  = (int*)      alloc((size_t)Bn * S1 * 4);
    float*     l1_xyz  = (float*)    alloc((size_t)Bn * S1 * 3 * 4);
    float*     l2_xyz  = (float*)    alloc((size_t)Bn * S2 * 3 * 4);
    int*       idxAr   = (int*)      alloc((size_t)Bn * S1 * 128 * 4);
    _Float16*  Wh      = (_Float16*) alloc((size_t)1024 * 1024 * 2);
    _Float16*  l1_feat = (_Float16*) alloc((size_t)Bn * S1 * C1 * 2);
    _Float16*  l2_feat = (_Float16*) alloc((size_t)Bn * S2 * C2 * 2);
    _Float16*  sa3feat = (_Float16*) alloc((size_t)16 * 1024 * 2);
    _Float16*  headtmp = (_Float16*) alloc((size_t)16 * 256 * 2);
    _Float16*  headout = (_Float16*) alloc((size_t)16 * 128 * 2);
    _Float16*  arenaA  = (_Float16*) alloc((size_t)524288 * 128 * 2);
    _Float16*  arenaB  = (_Float16*) alloc((size_t)524288 * 128 * 2);

    // ---- helpers ----
    auto run_mlp = [&](_Float16* cur, _Float16* nxt, int rows, int Kp,
                       const Layer* layers, int nl) -> std::pair<_Float16*, int> {
        for (int li = 0; li < nl; ++li) {
            const Layer& L = layers[li];
            int wn = L.O * Kp;
            conv_weight<<<div_up(wn, 256), 256, 0, stream>>>(L.W, Wh, L.O, L.C, Kp);
            dim3 g(div_up(rows, 128), L.O / 32);
            gemm_bn_relu<<<g, 256, 0, stream>>>(cur, Wh, L.b, L.g, L.be, nxt, rows, Kp, L.O);
            _Float16* t = cur; cur = nxt; nxt = t;
            Kp = L.O;
        }
        return {cur, Kp};   // final output buffer and its width
    };

    auto run_scale = [&](const float* src_xyz, int Nsrc, const float* centers, int S,
                         const _Float16* featIn, int Cfeat, float radius, int K,
                         const Layer* layers, int nl,
                         _Float16* featOut, int Ctot, int coff) {
        int ncent = Bn * S;
        int rows  = ncent * K;
        query_ball_kernel<<<div_up(ncent * 32, 256), 256, 0, stream>>>(
            src_xyz, centers, Nsrc, S, radius * radius, K, idxAr, ncent);
        int Kp0 = pad32(Cfeat + 3);
        if (Cfeat == 0) {
            gather_group_xyz<<<div_up(rows, 256), 256, 0, stream>>>(
                src_xyz, centers, idxAr, arenaA, Nsrc, S, K, rows);
        } else {
            gather_group_feat<<<div_up(rows * 32, 256), 256, 0, stream>>>(
                src_xyz, centers, featIn, idxAr, arenaA, Nsrc, S, K, Cfeat, Kp0, rows);
        }
        auto fin = run_mlp(arenaA, arenaB, rows, Kp0, layers, nl);
        int Olast = fin.second;
        pool_max_k<<<div_up(ncent * Olast, 256), 256, 0, stream>>>(
            fin.first, featOut, K, Olast, Ctot, coff, ncent);
    };

    // ---- stage 0: layout ----
    transpose_xyz<<<div_up(Bn * N, 256), 256, 0, stream>>>(input, xyz_t, Bn, N);

    // ---- SA1 (MSG) ----
    fps_kernel<<<Bn, 1024, 0, stream>>>(xyz_t, N, S1, fpsIdx);
    gather_centers<<<div_up(Bn * S1, 256), 256, 0, stream>>>(xyz_t, fpsIdx, l1_xyz, N, S1, Bn * S1);
    {
        int coff = 0;
        for (int s = 0; s < 3; ++s) {
            run_scale(xyz_t, N, l1_xyz, S1, nullptr, 0, r1[s], k1[s],
                      sa1[s], 3, l1_feat, C1, coff);
            coff += w1[s][2];
        }
    }

    // ---- SA2 (MSG) ----
    fps_kernel<<<Bn, 1024, 0, stream>>>(l1_xyz, S1, S2, fpsIdx);
    gather_centers<<<div_up(Bn * S2, 256), 256, 0, stream>>>(l1_xyz, fpsIdx, l2_xyz, S1, S2, Bn * S2);
    {
        int coff = 0;
        for (int s = 0; s < 3; ++s) {
            run_scale(l1_xyz, S1, l2_xyz, S2, l1_feat, C1, r2v[s], k2[s],
                      sa2[s], 3, l2_feat, C2, coff);
            coff += w2[s][2];
        }
    }

    // ---- SA3 (global) ----
    {
        int rows = Bn * S2;                 // 1024
        int Kp0  = pad32(C2 + 3);           // 672
        gather_sa3<<<div_up(rows * 32, 256), 256, 0, stream>>>(
            l2_xyz, l2_feat, arenaA, rows, C2, Kp0);
        auto fin = run_mlp(arenaA, arenaB, rows, Kp0, sa3, 3);   // -> [1024, 1024]
        zero_f16<<<div_up(16 * 1024, 256), 256, 0, stream>>>(sa3feat, 16 * 1024);
        pool_max_k<<<div_up(Bn * 1024, 256), 256, 0, stream>>>(
            fin.first, sa3feat, S2, 1024, 1024, 0, Bn);          // max over 128 pts
    }

    // ---- head: two FC + BN + ReLU (M=16 padded rows, batches in rows 0..7) ----
    {
        const Layer& L = head[0];
        conv_weight<<<div_up(L.O * 1024, 256), 256, 0, stream>>>(L.W, Wh, L.O, L.C, 1024);
        dim3 g(div_up(16, 128), L.O / 32);
        gemm_bn_relu<<<g, 256, 0, stream>>>(sa3feat, Wh, L.b, L.g, L.be, headtmp, 16, 1024, L.O);
    }
    {
        const Layer& L = head[1];
        conv_weight<<<div_up(L.O * 256, 256), 256, 0, stream>>>(L.W, Wh, L.O, L.C, 256);
        dim3 g(div_up(16, 128), L.O / 32);
        gemm_bn_relu<<<g, 256, 0, stream>>>(headtmp, Wh, L.b, L.g, L.be, headout, 16, 256, L.O);
    }

    // ---- output: (8,128,1) fp32 ----
    copy_out_f32<<<div_up(Bn * 128, 256), 256, 0, stream>>>(headout, (float*)d_out, Bn * 128);
}
